// Linear_Net_ASAP_72060961292411
// MI455X (gfx1250) — matrix operand order repaired and checked by reading, not compiled, not run on hardware
//
#include <hip/hip_runtime.h>
#include <hip/hip_bf16.h>

// ---------------------------------------------------------------------------
// Sizes (compile-time; match reference)
// ---------------------------------------------------------------------------
#define NNODES  65536      // B * NPG
#define NGRAPH  128
#define NPG     512
#define KSEL    256        // ratio 0.5
#define CHN     128        // IN_C == HID == OUT_C
#define NEGSL   0.2f
#define LCAP    64         // cached neighbor rows per node (deg avg ~17)
#define RT      4          // row tiles (of 16) per GEMM block

typedef __attribute__((ext_vector_type(16))) _Float16 v16h;
typedef __attribute__((ext_vector_type(8)))  _Float16 v8h;
typedef __attribute__((ext_vector_type(4)))  _Float16 v4h;
typedef __attribute__((ext_vector_type(8)))  float    v8f;

// ---------------------------------------------------------------------------
// Utility kernels
// ---------------------------------------------------------------------------
__global__ void zero_int_kernel(int* p, int n) {
    int i = blockIdx.x * blockDim.x + threadIdx.x;
    if (i < n) p[i] = 0;
}

// transpose + f32->f16 of a 128x128 weight (W[k][n] -> Wt[n][k])
__global__ void transpose_f16_kernel(const float* __restrict__ W,
                                     _Float16* __restrict__ Wt) {
    int idx = blockIdx.x * blockDim.x + threadIdx.x;   // 0..16383
    int k = idx >> 7, n = idx & 127;
    Wt[n * CHN + k] = (_Float16)W[k * CHN + n];
}

// wv[k] = sum_n Wp[k][n] * Wa_lo[n] ; qb = dot(bp, Wa_lo)
__global__ __launch_bounds__(128)
void prep_attn_kernel(const float* __restrict__ Wp, const float* __restrict__ bp,
                      const float* __restrict__ Wa, float* __restrict__ wv,
                      float* __restrict__ qb) {
    int k = threadIdx.x;
    float s = 0.f;
    for (int n = 0; n < CHN; ++n) s += Wp[k * CHN + n] * Wa[n];
    wv[k] = s;
    if (k == 0) {
        float q = 0.f;
        for (int n = 0; n < CHN; ++n) q += bp[n] * Wa[n];
        qb[0] = q;
    }
}

// ---------------------------------------------------------------------------
// WMMA GEMM: C[M][128] = act(A[M][128] @ B + bias), B transposed in f16.
// Block = 256 threads (8 waves). A 64-row tile is staged once in LDS (f16);
// each wave computes 4 row-tiles x one 16-col tile, reusing B fragments.
// A/B K order is contiguous (k = 16*half + s); C/D 32-bit 16x16 layout per CDNA5 ISA 7.12.2.
// ---------------------------------------------------------------------------
__global__ __launch_bounds__(256)
void wmma_gemm_kernel(const float* __restrict__ A, const _Float16* __restrict__ Bt,
                      const float* __restrict__ bias, float* __restrict__ C,
                      int relu) {
    __shared__ _Float16 at[RT * 16 * CHN];             // 64 x 128 f16 = 16 KB
    const int tid = threadIdx.x;

    // ---- stage A tile: 64 rows x 128 cols f32 -> f16 in LDS ----
    const float4* ag = (const float4*)(A + (size_t)blockIdx.x * (RT * 16) * CHN);
#pragma unroll
    for (int it = 0; it < 8; ++it) {                   // 2048 float4 / 256 thr
        int i = tid + it * 256;
        float4 v = ag[i];
        v4h hv = { (_Float16)v.x, (_Float16)v.y, (_Float16)v.z, (_Float16)v.w };
        *(v4h*)(at + i * 4) = hv;
    }
    __syncthreads();

    const int wid  = tid >> 5;
    const int lane = tid & 31;
    const int hb   = lane >> 4;        // lane half (0: lanes 0-15, 1: 16-31)
    const int lm   = lane & 15;
    const int c0   = wid * 16;         // column tile of this wave

    const _Float16* brow = Bt + (size_t)(c0 + lm) * CHN;

    v8f acc[RT] = {};
#pragma unroll
    for (int kk = 0; kk < CHN; kk += 32) {
        // B fragment (32x16): lane holds 16 contiguous K of one column
        v16h b = *(const v16h*)(brow + kk + hb * 16);
        const int o1 = kk + hb * 16;         // A K-offsets: 16 contiguous K, matching B
        const int o2 = kk + hb * 16 + 8;
#pragma unroll
        for (int rt = 0; rt < RT; ++rt) {
            const _Float16* arow = at + (rt * 16 + lm) * CHN;
            v16h a;
            *((v8h*)&a)     = *(const v8h*)(arow + o1);
            *((v8h*)&a + 1) = *(const v8h*)(arow + o2);
            acc[rt] = __builtin_amdgcn_wmma_f32_16x16x32_f16(
                          false, a, false, b, (short)0, acc[rt], false, false);
        }
    }

    const float bcol = bias[c0 + lm];
#pragma unroll
    for (int rt = 0; rt < RT; ++rt) {
#pragma unroll
        for (int r = 0; r < 8; ++r) {
            int row = blockIdx.x * (RT * 16) + rt * 16 + r + 8 * hb;
            float v = acc[rt][r] + bcol;
            if (relu) v = v > 0.f ? v : 0.f;
            C[(size_t)row * CHN + c0 + lm] = v;
        }
    }
}

// ---------------------------------------------------------------------------
// CSR construction (by dst)
// ---------------------------------------------------------------------------
__global__ void count_kernel(const long long* __restrict__ dst, int* cnt, int E) {
    int e = blockIdx.x * blockDim.x + threadIdx.x;
    if (e < E) atomicAdd(&cnt[(int)dst[e]], 1);
}

__global__ __launch_bounds__(1024)
void scan_kernel(const int* __restrict__ cnt, int* __restrict__ rowptr,
                 int* __restrict__ cursor) {
    __shared__ int sums[1024];
    const int t = threadIdx.x;
    const int per = NNODES / 1024;          // 64
    const int base = t * per;
    int s = 0;
    for (int i = 0; i < per; ++i) s += cnt[base + i];
    sums[t] = s;
    __syncthreads();
    for (int off = 1; off < 1024; off <<= 1) {
        int v = sums[t];
        int add = (t >= off) ? sums[t - off] : 0;
        __syncthreads();
        sums[t] = v + add;
        __syncthreads();
    }
    int run = sums[t] - s;                  // exclusive prefix
    for (int i = 0; i < per; ++i) {
        rowptr[base + i] = run;
        cursor[base + i] = run;
        run += cnt[base + i];
    }
    if (t == 1023) rowptr[NNODES] = run;
}

__global__ void scatter_kernel(const long long* __restrict__ src,
                               const long long* __restrict__ dst,
                               int* cursor, int* __restrict__ csr_src, int E) {
    int e = blockIdx.x * blockDim.x + threadIdx.x;
    if (e < E) {
        int d = (int)dst[e];
        int p = atomicAdd(&cursor[d], 1);
        csr_src[p] = (int)src[e];
    }
}

// ---------------------------------------------------------------------------
// Per-node scalar: h_s[j] = dot(h[j], Wa[128:256])
// ---------------------------------------------------------------------------
__global__ __launch_bounds__(128)
void node_hs_kernel(const float* __restrict__ h, const float* __restrict__ Wa,
                    float* __restrict__ h_s) {
    const int i = blockIdx.x, c = threadIdx.x;
    __shared__ float red[128];
    red[c] = h[(size_t)i * CHN + c] * Wa[CHN + c];
    __syncthreads();
    for (int off = 64; off > 0; off >>= 1) {
        if (c < off) red[c] += red[c + off];
        __syncthreads();
    }
    if (c == 0) h_s[i] = red[0];
}

// ---------------------------------------------------------------------------
// Fully fused per-node ASAP kernel:
//   - stage neighbor rows h[src] in LDS (read from HBM exactly once)
//   - segment-max -> x_q -> q_s[i] = x_q . wv + qb
//   - leaky-relu attention scores, per-node softmax (weights cached in LDS)
//   - weighted aggregation -> xc[i]
//   - LEConv dots -> a[i], fitbase[i]
// ---------------------------------------------------------------------------
__global__ __launch_bounds__(128)
void asap_node_kernel(const float* __restrict__ h, const float* __restrict__ h_s,
                      const int* __restrict__ rowptr, const int* __restrict__ csr_src,
                      const float* __restrict__ wv, const float* __restrict__ qb,
                      const float* __restrict__ ba,
                      const float* __restrict__ Wl1, const float* __restrict__ bl1,
                      const float* __restrict__ Wl2, const float* __restrict__ Wl3,
                      const float* __restrict__ bl3, float* __restrict__ xc,
                      float* __restrict__ a_out, float* __restrict__ fitbase) {
    __shared__ float rows[LCAP * CHN];      // 32 KB neighbor-row cache
    __shared__ float sc[LCAP];              // per-edge score -> weight
    __shared__ float red[CHN];

    const int i = blockIdx.x, c = threadIdx.x;
    const int beg = rowptr[i], end = rowptr[i + 1];
    const int deg = end - beg;
    const int ncache = deg < LCAP ? deg : LCAP;

    // ---- stage neighbor rows (one 512B coalesced row per iteration) ----
    for (int e = 0; e < ncache; ++e) {
        int j = csr_src[beg + e];
        if (e + 1 < ncache)                  // warm next row while this lands
            __builtin_prefetch(h + (size_t)csr_src[beg + e + 1] * CHN + c, 0, 0);
        rows[e * CHN + c] = h[(size_t)j * CHN + c];
    }
    __syncthreads();

    // ---- channel-wise segment max ----
    float m = -3.402823466e38f;
    for (int e = 0; e < ncache; ++e) m = fmaxf(m, rows[e * CHN + c]);
    for (int e = beg + LCAP; e < end; ++e) {            // rare overflow
        int j = csr_src[e];
        m = fmaxf(m, h[(size_t)j * CHN + c]);
    }

    // ---- q_s[i] = x_q . wv + qb ----
    red[c] = m * wv[c];
    __syncthreads();
    for (int off = 64; off > 0; off >>= 1) {
        if (c < off) red[c] += red[c + off];
        __syncthreads();
    }
    const float qsi = red[0] + qb[0];
    __syncthreads();

    // ---- scores (strided over edges), cache, track max ----
    const float ba0 = ba[0];
    float lmax = -3.402823466e38f;
    for (int e = c; e < deg; e += 128) {
        float s = qsi + h_s[csr_src[beg + e]] + ba0;
        s = s > 0.f ? s : NEGSL * s;
        if (e < LCAP) sc[e] = s;
        lmax = fmaxf(lmax, s);
    }
    red[c] = lmax;
    __syncthreads();
    for (int off = 64; off > 0; off >>= 1) {
        if (c < off) red[c] = fmaxf(red[c], red[c + off]);
        __syncthreads();
    }
    const float smax = red[0];
    __syncthreads();

    // ---- convert cached scores to weights; partial partition sum ----
    float ps = 0.f;
    for (int e = c; e < deg; e += 128) {
        float w;
        if (e < LCAP) {
            w = __expf(sc[e] - smax);
            sc[e] = w;
        } else {
            float s = qsi + h_s[csr_src[beg + e]] + ba0;
            s = s > 0.f ? s : NEGSL * s;
            w = __expf(s - smax);
        }
        ps += w;
    }
    red[c] = ps;
    __syncthreads();
    for (int off = 64; off > 0; off >>= 1) {
        if (c < off) red[c] += red[c + off];
        __syncthreads();
    }
    const float ssum = red[0];
    __syncthreads();

    // ---- weighted aggregation from the LDS cache ----
    float acc = 0.f;
    for (int e = 0; e < ncache; ++e) acc += sc[e] * rows[e * CHN + c];
    for (int e = beg + LCAP; e < end; ++e) {            // rare overflow
        int j = csr_src[e];
        float s = qsi + h_s[j] + ba0;
        s = s > 0.f ? s : NEGSL * s;
        acc += __expf(s - smax) * h[(size_t)j * CHN + c];
    }
    const float xcv = acc / ssum;
    xc[(size_t)i * CHN + c] = xcv;

    // ---- LEConv dots: a = xc.Wl1+bl1, bb = xc.Wl2, l3 = xc.Wl3+bl3 ----
    red[c] = xcv * Wl1[c];
    __syncthreads();
    for (int off = 64; off > 0; off >>= 1) {
        if (c < off) red[c] += red[c + off];
        __syncthreads();
    }
    const float d1 = red[0];
    __syncthreads();
    red[c] = xcv * Wl2[c];
    __syncthreads();
    for (int off = 64; off > 0; off >>= 1) {
        if (c < off) red[c] += red[c + off];
        __syncthreads();
    }
    const float d2 = red[0];
    __syncthreads();
    red[c] = xcv * Wl3[c];
    __syncthreads();
    for (int off = 64; off > 0; off >>= 1) {
        if (c < off) red[c] += red[c + off];
        __syncthreads();
    }
    const float d3 = red[0];
    if (c == 0) {
        float dg = (float)deg;
        a_out[i]   = d1 + bl1[0];
        fitbase[i] = d3 + bl3[0] - dg * d2;
    }
}

// ---------------------------------------------------------------------------
// fit[i] = sigmoid(sum_{j in N(i)} a[j] + fitbase[i])
// ---------------------------------------------------------------------------
__global__ void fit_kernel(const int* __restrict__ rowptr, const int* __restrict__ csr_src,
                           const float* __restrict__ a, const float* __restrict__ fitbase,
                           float* __restrict__ fit) {
    int i = blockIdx.x * blockDim.x + threadIdx.x;
    if (i >= NNODES) return;
    int beg = rowptr[i], end = rowptr[i + 1];
    float s = 0.f;
    for (int e = beg; e < end; ++e) s += a[csr_src[e]];
    float f = s + fitbase[i];
    fit[i] = 1.f / (1.f + __expf(-f));
}

// ---------------------------------------------------------------------------
// Per-graph top-K select (rank-by-comparison) fused with the pooled sum:
// zg[g] = (1/K) * sum_{i in topK(g)} fit[i] * xc[i]
// ---------------------------------------------------------------------------
__global__ __launch_bounds__(512)
void topk_accum_kernel(const float* __restrict__ fit, const float* __restrict__ xc,
                       float* __restrict__ zg) {
    const int g = blockIdx.x, t = threadIdx.x;
    __shared__ float f[NPG];
    __shared__ float z[CHN];
    f[t] = fit[g * NPG + t];
    if (t < CHN) z[t] = 0.f;
    __syncthreads();
    const float fi = f[t];
    int rank = 0;
    for (int j = 0; j < NPG; ++j) {
        float fj = f[j];
        rank += (fj > fi) || (fj == fi && j < t);   // stable: lower index wins ties
    }
    if (rank < KSEL) {
        const float* row = xc + (size_t)(g * NPG + t) * CHN;
        for (int k = 0; k < CHN; ++k) {
            int c = (t + k) & (CHN - 1);            // rotate to spread LDS banks
            atomicAdd(&z[c], fi * row[c]);
        }
    }
    __syncthreads();
    if (t < CHN) zg[g * CHN + t] = z[t] * (1.0f / (float)KSEL);
}

// ---------------------------------------------------------------------------
// Host-side orchestration
// ---------------------------------------------------------------------------
static inline size_t alignup(size_t x) { return (x + 255) & ~(size_t)255; }

extern "C" void kernel_launch(void* const* d_in, const int* in_sizes, int n_in,
                              void* d_out, int out_size, void* d_ws, size_t ws_size,
                              hipStream_t stream) {
    const float* x   = (const float*)d_in[0];
    const float* W1  = (const float*)d_in[1];
    const float* b1  = (const float*)d_in[2];
    const float* Wp  = (const float*)d_in[3];
    const float* bp  = (const float*)d_in[4];
    const float* Wa  = (const float*)d_in[5];
    const float* ba  = (const float*)d_in[6];
    const float* Wl1 = (const float*)d_in[7];
    const float* bl1 = (const float*)d_in[8];
    const float* Wl2 = (const float*)d_in[9];
    const float* Wl3 = (const float*)d_in[10];
    const float* bl3 = (const float*)d_in[11];
    const float* W2  = (const float*)d_in[12];
    const float* b2  = (const float*)d_in[13];
    const long long* ei = (const long long*)d_in[14];
    // d_in[15] (batch) unused: graphs are contiguous blocks of NPG nodes.

    const int E = in_sizes[14] / 2;            // 1,114,112
    const long long* srcp = ei;
    const long long* dstp = ei + E;

    // workspace carve-out
    char* w = (char*)d_ws;
    size_t off = 0;
    float*    h       = (float*)(w + off); off += alignup((size_t)NNODES * CHN * 4);
    float*    xc      = (float*)(w + off); off += alignup((size_t)NNODES * CHN * 4);
    int*      csr_src = (int*)(w + off);   off += alignup((size_t)E * 4);
    int*      rowptr  = (int*)(w + off);   off += alignup((size_t)(NNODES + 1) * 4);
    int*      cursor  = (int*)(w + off);   off += alignup((size_t)NNODES * 4);
    int*      cnt     = (int*)(w + off);   off += alignup((size_t)NNODES * 4);
    float*    h_s     = (float*)(w + off); off += alignup((size_t)NNODES * 4);
    float*    a_arr   = (float*)(w + off); off += alignup((size_t)NNODES * 4);
    float*    fitbase = (float*)(w + off); off += alignup((size_t)NNODES * 4);
    float*    fit     = (float*)(w + off); off += alignup((size_t)NNODES * 4);
    float*    wv      = (float*)(w + off); off += alignup((size_t)CHN * 4);
    float*    qb      = (float*)(w + off); off += alignup(256);
    _Float16* W1t     = (_Float16*)(w + off); off += alignup((size_t)CHN * CHN * 2);
    _Float16* W2t     = (_Float16*)(w + off); off += alignup((size_t)CHN * CHN * 2);
    float*    zg      = (float*)(w + off); off += alignup((size_t)NGRAPH * CHN * 4);
    (void)ws_size; (void)n_in; (void)out_size;

    // 0) init + weight prep
    zero_int_kernel<<<NNODES / 256, 256, 0, stream>>>(cnt, NNODES);
    transpose_f16_kernel<<<64, 256, 0, stream>>>(W1, W1t);
    transpose_f16_kernel<<<64, 256, 0, stream>>>(W2, W2t);
    prep_attn_kernel<<<1, 128, 0, stream>>>(Wp, bp, Wa, wv, qb);

    // 1) h = relu(x @ W1 + b1)   (WMMA, LDS-staged A, 64 rows/block)
    wmma_gemm_kernel<<<NNODES / (RT * 16), 256, 0, stream>>>(x, W1t, b1, h, 1);

    // 2) per-node attention scalar h_s
    node_hs_kernel<<<NNODES, 128, 0, stream>>>(h, Wa, h_s);

    // 3) CSR by dst
    count_kernel<<<(E + 255) / 256, 256, 0, stream>>>(dstp, cnt, E);
    scan_kernel<<<1, 1024, 0, stream>>>(cnt, rowptr, cursor);
    scatter_kernel<<<(E + 255) / 256, 256, 0, stream>>>(srcp, dstp, cursor, csr_src, E);

    // 4) fused segment-max + attention softmax + aggregation + LEConv dots
    asap_node_kernel<<<NNODES, 128, 0, stream>>>(h, h_s, rowptr, csr_src,
                                                 wv, qb, ba, Wl1, bl1, Wl2, Wl3, bl3,
                                                 xc, a_arr, fitbase);

    // 5) LEConv fitness + sigmoid
    fit_kernel<<<NNODES / 256, 256, 0, stream>>>(rowptr, csr_src, a_arr, fitbase, fit);

    // 6) per-graph top-K selection fused with pooled weighted sum
    topk_accum_kernel<<<NGRAPH, NPG, 0, stream>>>(fit, xc, zg);

    // 7) out = zg @ W2 + b2   (WMMA, 128x128x128)
    wmma_gemm_kernel<<<NGRAPH / (RT * 16), 256, 0, stream>>>(zg, W2t, b2,
                                                             (float*)d_out, 0);
}
